// MambaTemporal_34883724378455
// MI455X (gfx1250) — compile-verified, matrix operand order checked by reading, not run on hardware
//
#include <hip/hip_runtime.h>
#include <hip/hip_bf16.h>

// ---------------------------------------------------------------------------
// Bidirectional Mamba block for gfx1250 (MI455X), wave32 + WMMA bf16.
// GEMMs consume pre-converted bf16 operands and stream tiles into LDS with
// GLOBAL_LOAD_ASYNC_TO_LDS_B128 (double-buffered, ASYNCcnt-synchronized),
// falling back to sync b128 copies if the async builtin is unavailable.
// ---------------------------------------------------------------------------

typedef __bf16 bf16_t;
typedef __attribute__((ext_vector_type(16))) __bf16 v16bf;
typedef __attribute__((ext_vector_type(8)))  float  v8f;
typedef __attribute__((ext_vector_type(4)))  int    v4i;

#define LDS_STRIDE 40  // 32 K-values + 8 bf16 pad (80B rows, 16B aligned)

#if defined(__has_builtin)
#if __has_builtin(__builtin_amdgcn_global_load_async_to_lds_b128)
#define USE_ASYNC 1
#else
#define USE_ASYNC 0
#endif
#else
#define USE_ASYNC 0
#endif

typedef __attribute__((address_space(1))) v4i gv4i_t;  // global int4
typedef __attribute__((address_space(3))) v4i lv4i_t;  // LDS int4

__device__ __forceinline__ void wait_async0() {
#if __has_builtin(__builtin_amdgcn_s_wait_asynccnt)
  __builtin_amdgcn_s_wait_asynccnt(0);
#else
  asm volatile("s_wait_asynccnt 0" ::: "memory");
#endif
}

#if USE_ASYNC
__device__ __forceinline__ void async_b128(const bf16_t* g, bf16_t* l) {
  __builtin_amdgcn_global_load_async_to_lds_b128(
      (gv4i_t*)(void*)g, (lv4i_t*)l, 0, 0);
}
#endif

// Per-lane 16x32 bf16 fragment from a row-major LDS tile (stride LDS_STRIDE).
// ISA layout: row = lane&15; K-chunk base = (lane>>4)*8, second chunk +16.
__device__ __forceinline__ v16bf load_frag(const bf16_t* base, int lane) {
  const bf16_t* p = base + (lane & 15) * LDS_STRIDE + ((lane >> 4) << 3);
  union { v16bf v; float4 q[2]; } u;
  u.q[0] = *(const float4*)(p);
  u.q[1] = *(const float4*)(p + 16);
  return u.v;
}

// C = act(A[MxK]bf16 @ B[NxK]bf16^T + bias), written to f32 C32 and/or bf16
// C16.  M fixed = 8192 (grid.y = 64).  act: 0 none, 1 relu, 2 softplus.
__global__ void __launch_bounds__(256) gemm_bf16_wmma(
    const bf16_t* __restrict__ A, int lda,
    const bf16_t* __restrict__ B, int ldb,
    float* __restrict__ C32, int ldc32,
    bf16_t* __restrict__ C16, int ldc16,
    const float* __restrict__ bias,
    int N, int K, int act)
{
  __shared__ bf16_t As[2][128 * LDS_STRIDE];
  __shared__ bf16_t Bs[2][128 * LDS_STRIDE];
  const int tid  = threadIdx.x;
  const int lane = tid & 31;
  const int wave = tid >> 5;
  const int wm   = wave & 1;   // 2 wave-rows  (64 M each)
  const int wn   = wave >> 1;  // 4 wave-cols  (32 N each)
  const long blockM = (long)blockIdx.y * 128;
  const int  blockN = blockIdx.x * 128;

  // Per-thread tile-copy coordinates: tile = 128 rows x 32 bf16 = 512 x 16B
  // chunks; each thread owns 2 chunks of A and 2 of B per K-step.
  const int i0 = tid, i1 = tid + 256;
  const int ar0 = i0 >> 2, ac0 = (i0 & 3) << 3;
  const int ar1 = i1 >> 2, ac1 = (i1 & 3) << 3;
  const bf16_t* Ag0 = A + (blockM + ar0) * (long)lda + ac0;
  const bf16_t* Ag1 = A + (blockM + ar1) * (long)lda + ac1;
  const int nb0 = (blockN + ar0 < N) ? (blockN + ar0) : (N - 1);  // clamp:
  const int nb1 = (blockN + ar1 < N) ? (blockN + ar1) : (N - 1);  // masked
  const bf16_t* Bg0 = B + (long)nb0 * ldb + ac0;                  // in store
  const bf16_t* Bg1 = B + (long)nb1 * ldb + ac1;
  const int al0 = ar0 * LDS_STRIDE + ac0;
  const int al1 = ar1 * LDS_STRIDE + ac1;

  const v8f vzero = {0.f, 0.f, 0.f, 0.f, 0.f, 0.f, 0.f, 0.f};
  v8f acc[4][2];
#pragma unroll
  for (int i = 0; i < 4; ++i)
#pragma unroll
    for (int j = 0; j < 2; ++j) acc[i][j] = vzero;

#if USE_ASYNC
  // ---- double-buffered async pipeline ----
  async_b128(Ag0, &As[0][al0]);
  async_b128(Ag1, &As[0][al1]);
  async_b128(Bg0, &Bs[0][al0]);
  async_b128(Bg1, &Bs[0][al1]);
  int buf = 0;
  for (int k0 = 0; k0 < K; k0 += 32, buf ^= 1) {
    wait_async0();        // this wave's tile-k writes have landed in LDS
    __syncthreads();      // everyone's have; everyone done reading buf^1
    if (k0 + 32 < K) {
      const int nk = k0 + 32;
      async_b128(Ag0 + nk, &As[buf ^ 1][al0]);
      async_b128(Ag1 + nk, &As[buf ^ 1][al1]);
      async_b128(Bg0 + nk, &Bs[buf ^ 1][al0]);
      async_b128(Bg1 + nk, &Bs[buf ^ 1][al1]);
    }
    v16bf af[4], bfg[2];
#pragma unroll
    for (int i = 0; i < 4; ++i)
      af[i] = load_frag(&As[buf][(wm * 64 + i * 16) * LDS_STRIDE], lane);
#pragma unroll
    for (int j = 0; j < 2; ++j)
      bfg[j] = load_frag(&Bs[buf][(wn * 32 + j * 16) * LDS_STRIDE], lane);
#pragma unroll
    for (int i = 0; i < 4; ++i)
#pragma unroll
      for (int j = 0; j < 2; ++j)
        acc[i][j] = __builtin_amdgcn_wmma_f32_16x16x32_bf16(
            false, af[i], false, bfg[j], (short)0, acc[i][j], false, false);
  }
#else
  // ---- synchronous fallback: b128 load -> ds_store_b128, single buffer ----
  for (int k0 = 0; k0 < K; k0 += 32) {
    *(float4*)&As[0][al0] = *(const float4*)(Ag0 + k0);
    *(float4*)&As[0][al1] = *(const float4*)(Ag1 + k0);
    *(float4*)&Bs[0][al0] = *(const float4*)(Bg0 + k0);
    *(float4*)&Bs[0][al1] = *(const float4*)(Bg1 + k0);
    __syncthreads();
    v16bf af[4], bfg[2];
#pragma unroll
    for (int i = 0; i < 4; ++i)
      af[i] = load_frag(&As[0][(wm * 64 + i * 16) * LDS_STRIDE], lane);
#pragma unroll
    for (int j = 0; j < 2; ++j)
      bfg[j] = load_frag(&Bs[0][(wn * 32 + j * 16) * LDS_STRIDE], lane);
#pragma unroll
    for (int i = 0; i < 4; ++i)
#pragma unroll
      for (int j = 0; j < 2; ++j)
        acc[i][j] = __builtin_amdgcn_wmma_f32_16x16x32_bf16(
            false, af[i], false, bfg[j], (short)0, acc[i][j], false, false);
    __syncthreads();
  }
#endif

  // ---- epilogue: bias + activation, scatter per ISA C/D layout ----
  const long rbase = blockM + wm * 64;
  const int  cbase = blockN + wn * 32;
  const int  cl = lane & 15;
  const int  rh = (lane >> 4) * 8;
#pragma unroll
  for (int i = 0; i < 4; ++i) {
#pragma unroll
    for (int j = 0; j < 2; ++j) {
      int col = cbase + j * 16 + cl;
      if (col < N) {
        float bv = bias ? bias[col] : 0.f;
#pragma unroll
        for (int r = 0; r < 8; ++r) {
          float v = acc[i][j][r] + bv;
          if (act == 1)      v = fmaxf(v, 0.f);
          else if (act == 2) v = (v > 20.f) ? v : logf(1.f + __expf(v));
          long row = rbase + i * 16 + rh + r;
          if (C32) C32[row * (long)ldc32 + col] = v;
          if (C16) C16[row * (long)ldc16 + col] = (bf16_t)v;
        }
      }
    }
  }
}

// Elementwise f32 -> bf16 (weights / input pre-conversion).
__global__ void __launch_bounds__(256) cvt_f32_to_bf16(
    const float* __restrict__ in, bf16_t* __restrict__ out, long n)
{
  long stride = (long)gridDim.x * 256;
  for (long i = (long)blockIdx.x * 256 + threadIdx.x; i < n; i += stride)
    out[i] = (bf16_t)in[i];
}

// Depthwise causal conv over time + SiLU.  xz holds xi in cols [0,2048) of a
// 4096-wide row.  reverse=1 implements the conv of the time-reversed sequence
// indexed in original time (anti-causal taps), so no physical flip is needed.
__global__ void __launch_bounds__(256) conv_silu_kernel(
    const float* __restrict__ xz, const float* __restrict__ cw,
    const float* __restrict__ cb, float* __restrict__ xc,
    bf16_t* __restrict__ xcbf, int reverse)
{
  long idx = (long)blockIdx.x * 256 + threadIdx.x;   // 8*1024*2048 threads
  int d = (int)(idx & 2047);
  int t = (int)((idx >> 11) & 1023);
  int b = (int)(idx >> 21);
  float a = cb[d];
#pragma unroll
  for (int k = 0; k < 4; ++k) {
    int tt = reverse ? (t + 3 - k) : (t - 3 + k);
    if (tt >= 0 && tt < 1024)
      a += cw[d * 4 + k] * xz[((long)b * 1024 + tt) * 4096 + d];
  }
  float s = a / (1.f + __expf(-a));   // silu
  xc[idx] = s;
  xcbf[idx] = (bf16_t)s;
}

__global__ void __launch_bounds__(256) aneg_kernel(
    const float* __restrict__ A_log, float* __restrict__ Aneg)
{
  int i = blockIdx.x * 256 + threadIdx.x;
  if (i < 2048 * 16) Aneg[i] = -__expf(A_log[i]);
}

// Selective scan: one lane per (batch, channel); 16 states in registers.
// Emits gated y directly as bf16 (only the out-proj GEMM consumes it).
__global__ void __launch_bounds__(256) mamba_scan_kernel(
    const float* __restrict__ delta,   // (8192,2048)
    const float* __restrict__ xdbl,    // (8192,96): B at +64, C at +80
    const float* __restrict__ xc,      // (8192,2048)
    const float* __restrict__ xz,      // (8192,4096): z at col 2048+d
    const float* __restrict__ Aneg,    // (2048,16)
    const float* __restrict__ Dp,      // (2048,)
    bf16_t* __restrict__ ybf,          // (8192,2048)
    int reverse)
{
  int gid = blockIdx.x * 256 + threadIdx.x;  // 0..16383
  int b = gid >> 11;
  int d = gid & 2047;
  float A[16], h[16];
#pragma unroll
  for (int s = 0; s < 16; ++s) { A[s] = Aneg[d * 16 + s]; h[s] = 0.f; }
  const float Dv = Dp[d];

  for (int step = 0; step < 1024; ++step) {
    int t = reverse ? (1023 - step) : step;
    long row = (long)b * 1024 + t;
    float dt  = delta[row * 2048 + d];
    float xct = xc[row * 2048 + d];
    float zt  = xz[row * 4096 + 2048 + d];
    float dx  = dt * xct;
    float y = 0.f;
#pragma unroll
    for (int s = 0; s < 16; ++s) {
      float Bs = xdbl[row * 96 + 64 + s];
      float Cs = xdbl[row * 96 + 80 + s];
      h[s] = __expf(dt * A[s]) * h[s] + dx * Bs;
      y = fmaf(h[s], Cs, y);
    }
    y = fmaf(xct, Dv, y);
    float gate = zt / (1.f + __expf(-zt));
    ybf[row * 2048 + d] = (bf16_t)(y * gate);
  }
}

static inline int cvt_grid(long n) {
  long g = (n + 255) / 256;
  return (int)(g < 4096 ? g : 4096);
}

extern "C" void kernel_launch(void* const* d_in, const int* in_sizes, int n_in,
                              void* d_out, int out_size, void* d_ws, size_t ws_size,
                              hipStream_t stream) {
  const float* x  = (const float*)d_in[0];   // (8,1024,1408)
  const float* W1 = (const float*)d_in[1];   // (1024,1408)
  const float* b1 = (const float*)d_in[2];   // (1024,)
  const float* W2 = (const float*)d_in[3];   // (72,2048)
  const float* b2 = (const float*)d_in[4];   // (72,)

  // Workspace carve (256B aligned)
  char* p = (char*)d_ws;
  auto carve = [&](size_t bytes) {
    char* r = p; p += (bytes + 255) & ~(size_t)255; return r;
  };
  float*  xz      = (float*) carve(8192ull * 4096 * 4);
  float*  xc      = (float*) carve(8192ull * 2048 * 4);
  float*  xdbl    = (float*) carve(8192ull * 96 * 4);
  float*  delta   = (float*) carve(8192ull * 2048 * 4);
  float*  aneg    = (float*) carve(2048ull * 16 * 4);
  bf16_t* x_bf    = (bf16_t*)carve(8192ull * 1408 * 2);
  bf16_t* h_bf    = (bf16_t*)carve(8192ull * 1024 * 2);
  bf16_t* xc_bf   = (bf16_t*)carve(8192ull * 2048 * 2);
  bf16_t* xdbl_bf = (bf16_t*)carve(8192ull * 96 * 2);
  bf16_t* y_bf    = (bf16_t*)carve(8192ull * 2048 * 2);
  bf16_t* ycat_bf = (bf16_t*)carve(8192ull * 2048 * 2);
  bf16_t* W1_bf   = (bf16_t*)carve(1024ull * 1408 * 2);
  bf16_t* W2_bf   = (bf16_t*)carve(72ull * 2048 * 2);
  bf16_t* inw_bf  = (bf16_t*)carve(4096ull * 1024 * 2);
  bf16_t* xpw_bf  = (bf16_t*)carve(96ull * 2048 * 2);
  bf16_t* dtw_bf  = (bf16_t*)carve(2048ull * 64 * 2);
  bf16_t* outw_bf = (bf16_t*)carve(1024ull * 2048 * 2);

  const dim3 blk(256);

  // One-time conversions
  cvt_f32_to_bf16<<<cvt_grid(8192ll * 1408), blk, 0, stream>>>(x,  x_bf,  8192ll * 1408);
  cvt_f32_to_bf16<<<cvt_grid(1024ll * 1408), blk, 0, stream>>>(W1, W1_bf, 1024ll * 1408);
  cvt_f32_to_bf16<<<cvt_grid(72ll * 2048),   blk, 0, stream>>>(W2, W2_bf, 72ll * 2048);

  // h = relu(x @ W1^T + b1)        M=8192 K=1408 N=1024  -> bf16 only
  gemm_bf16_wmma<<<dim3(8, 64), blk, 0, stream>>>(
      x_bf, 1408, W1_bf, 1408, nullptr, 0, h_bf, 1024, b1, 1024, 1408, 1);

  for (int dir = 0; dir < 2; ++dir) {
    const float* in_w    = (const float*)d_in[5 + 9 * dir];
    const float* conv_w  = (const float*)d_in[6 + 9 * dir];
    const float* conv_b  = (const float*)d_in[7 + 9 * dir];
    const float* xproj_w = (const float*)d_in[8 + 9 * dir];
    const float* dt_w    = (const float*)d_in[9 + 9 * dir];
    const float* dt_b    = (const float*)d_in[10 + 9 * dir];
    const float* A_log   = (const float*)d_in[11 + 9 * dir];
    const float* Dpp     = (const float*)d_in[12 + 9 * dir];
    const float* out_w   = (const float*)d_in[13 + 9 * dir];

    cvt_f32_to_bf16<<<cvt_grid(4096ll * 1024), blk, 0, stream>>>(in_w,    inw_bf,  4096ll * 1024);
    cvt_f32_to_bf16<<<cvt_grid(96ll * 2048),   blk, 0, stream>>>(xproj_w, xpw_bf,  96ll * 2048);
    cvt_f32_to_bf16<<<cvt_grid(2048ll * 64),   blk, 0, stream>>>(dt_w,    dtw_bf,  2048ll * 64);
    cvt_f32_to_bf16<<<cvt_grid(1024ll * 2048), blk, 0, stream>>>(out_w,   outw_bf, 1024ll * 2048);

    // xz = h @ in_w^T              M=8192 K=1024 N=4096  -> f32
    gemm_bf16_wmma<<<dim3(32, 64), blk, 0, stream>>>(
        h_bf, 1024, inw_bf, 1024, xz, 4096, nullptr, 0, nullptr, 4096, 1024, 0);

    // xc = silu(depthwise causal conv(xi))  -> f32 + bf16
    conv_silu_kernel<<<dim3(65536), blk, 0, stream>>>(
        xz, conv_w, conv_b, xc, xc_bf, dir);

    // x_dbl = xc @ xproj_w^T       M=8192 K=2048 N=96    -> f32 + bf16
    gemm_bf16_wmma<<<dim3(1, 64), blk, 0, stream>>>(
        xc_bf, 2048, xpw_bf, 2048, xdbl, 96, xdbl_bf, 96, nullptr, 96, 2048, 0);

    // delta = softplus(dt @ dt_w^T + dt_b)  M=8192 K=64 N=2048 -> f32
    gemm_bf16_wmma<<<dim3(16, 64), blk, 0, stream>>>(
        xdbl_bf, 96, dtw_bf, 64, delta, 2048, nullptr, 0, dt_b, 2048, 64, 2);

    // A = -exp(A_log)
    aneg_kernel<<<dim3(128), blk, 0, stream>>>(A_log, aneg);

    // selective scan + skip + gate -> y bf16
    mamba_scan_kernel<<<dim3(64), blk, 0, stream>>>(
        delta, xdbl, xc, xz, aneg, Dpp, y_bf, dir);

    // y_dir = y @ out_w^T -> ycat columns [dir*1024, dir*1024+1024)  (bf16)
    gemm_bf16_wmma<<<dim3(8, 64), blk, 0, stream>>>(
        y_bf, 2048, outw_bf, 2048, nullptr, 0, ycat_bf + (long)dir * 1024,
        2048, nullptr, 1024, 2048, 0);
  }

  // out = ycat @ W2^T + b2         M=8192 K=2048 N=72 -> f32 d_out
  gemm_bf16_wmma<<<dim3(1, 64), blk, 0, stream>>>(
      ycat_bf, 2048, W2_bf, 2048, (float*)d_out, 72, nullptr, 0, b2, 72, 2048, 0);
}
